// Model_53283364274775
// MI455X (gfx1250) — matrix operand order repaired and checked by reading, not compiled, not run on hardware
//
#include <hip/hip_runtime.h>

// ---------------------------------------------------------------------------
// MI455X / gfx1250, wave32. Matrix math via v_wmma_f32_16x16x32_f16.
//
// Kernel 1 (ui):  ui = LReLU(LReLU([u|i] @ W0 + b0) @ W1 + b1)   -> d_ws
// Kernel 2 (ao):  1024 blocks; each stages ALL 3 relations' weights
//                 (transposed f16) in LDS once (~146 KB), then loops over 4
//                 row-tiles of 64 rows: stage X=[a|o] (f16, b128 loads),
//                 prefetch next tile, run the 2-layer MLP for r=0..2 with
//                 WMMA (inline-0 C, bias folded into epilogue), fuse
//                 dot(ui, aos), commit pred[row] where s[row]==r.
// ---------------------------------------------------------------------------

typedef __attribute__((ext_vector_type(16))) _Float16 v16h;
typedef __attribute__((ext_vector_type(8)))  _Float16 v8h;
typedef __attribute__((ext_vector_type(4)))  _Float16 v4h;
typedef __attribute__((ext_vector_type(8)))  float    v8f;

#define SLOPE 0.01f
#define LREL(x) ((x) >= 0.0f ? (x) : SLOPE * (x))

static __device__ __forceinline__ v16h cat8(v8h lo, v8h hi) {
  return __builtin_shufflevector(lo, hi, 0,1,2,3,4,5,6,7,8,9,10,11,12,13,14,15);
}

static __device__ __forceinline__ v8f wmma16(v16h a, v16h b, v8f c) {
  // (neg_a, A, neg_b, B, c_mod, C, reuse_a, reuse_b)
  return __builtin_amdgcn_wmma_f32_16x16x32_f16(false, a, false, b, (short)0, c,
                                                false, false);
}

static __device__ __forceinline__ v4h cvt4(float4 v) {
  v4h h = {(_Float16)v.x, (_Float16)v.y, (_Float16)v.z, (_Float16)v.w};
  return h;
}

// 16-bit A 16x32 fragment from an LDS row (row-major, 128-half stride):
// lane (lq=lane&15 -> M, hi=lane>>4): v0..7 = K kbase+(hi?16:0)..+15
static __device__ __forceinline__ v16h load_a_frag(const _Float16* rowp, int k,
                                                   int hi) {
  const int off = 32 * k + (hi ? 16 : 0);
  v8h lo = *(const v8h*)(rowp + off);
  v8h hh = *(const v8h*)(rowp + off + 8);
  return cat8(lo, hh);
}

// B 32x16 fragment from transposed weights WT[out][in]:
// lane N = lane&15 selects the out-row; 16 contiguous K at 32k+(hi?16:0).
static __device__ __forceinline__ v16h load_b_frag(const _Float16* outrow,
                                                   int k, int hi) {
  return *(const v16h*)(outrow + 32 * k + (hi ? 16 : 0));
}

// ------------------------------- UI kernel ---------------------------------
__global__ __launch_bounds__(128) void ui_mlp_kernel(
    const float* __restrict__ u_emb, const float* __restrict__ i_emb,
    const float* __restrict__ W0, const float* __restrict__ b0,
    const float* __restrict__ W1, const float* __restrict__ b1,
    float* __restrict__ ui_out) {
  __shared__ __attribute__((aligned(32))) _Float16 Xs[64][128];
  __shared__ __attribute__((aligned(32))) _Float16 Hs[4][16][128];
  __shared__ __attribute__((aligned(32))) _Float16 W0T[128][128];
  __shared__ __attribute__((aligned(32))) _Float16 W1T[64][128];
  __shared__ float b0s[128];
  __shared__ float b1s[64];

  const int tid  = threadIdx.x;
  const int wave = tid >> 5;
  const int lane = tid & 31;
  const int lq   = lane & 15;
  const int hi   = lane >> 4;
  const int rowBase = blockIdx.x * 64;

  // stage X = [u | i] as f16 (b128 global loads, b64 LDS stores)
  const float4* u4 = (const float4*)u_emb;
  const float4* i4 = (const float4*)i_emb;
  for (int idx = tid; idx < 64 * 16; idx += 128) {
    const int row = idx >> 4, q = idx & 15;
    const int g = (rowBase + row) * 16 + q;
    *(v4h*)&Xs[row][q * 4]      = cvt4(u4[g]);
    *(v4h*)&Xs[row][64 + q * 4] = cvt4(i4[g]);
  }
  // stage transposed weights + biases (b128 global loads)
  const float4* W0v = (const float4*)W0;
  for (int idx = tid; idx < (128 * 128) / 4; idx += 128) {
    const int e = idx * 4;
    const int i = e >> 7, o = e & 127;
    const float4 w = W0v[idx];
    W0T[o + 0][i] = (_Float16)w.x;
    W0T[o + 1][i] = (_Float16)w.y;
    W0T[o + 2][i] = (_Float16)w.z;
    W0T[o + 3][i] = (_Float16)w.w;
  }
  const float4* W1v = (const float4*)W1;
  for (int idx = tid; idx < (128 * 64) / 4; idx += 128) {
    const int e = idx * 4;
    const int i = e >> 6, o = e & 63;
    const float4 w = W1v[idx];
    W1T[o + 0][i] = (_Float16)w.x;
    W1T[o + 1][i] = (_Float16)w.y;
    W1T[o + 2][i] = (_Float16)w.z;
    W1T[o + 3][i] = (_Float16)w.w;
  }
  if (tid < 128) b0s[tid] = b0[tid];
  if (tid < 64)  b1s[tid] = b1[tid];
  __syncthreads();

  const int wrow = rowBase + wave * 16;
  const _Float16* xrow = &Xs[wave * 16 + lq][0];

  v16h aX[4];
#pragma unroll
  for (int k = 0; k < 4; ++k) aX[k] = load_a_frag(xrow, k, hi);

#pragma unroll
  for (int n = 0; n < 8; ++n) {
    v8f c = {};  // folds to inline-0 SRC2 on the first WMMA
#pragma unroll
    for (int k = 0; k < 4; ++k)
      c = wmma16(aX[k], load_b_frag(&W0T[n * 16 + lq][0], k, hi), c);
    const float bn = b0s[n * 16 + lq];
#pragma unroll
    for (int i = 0; i < 8; ++i) {
      const float h = LREL(c[i] + bn);
      Hs[wave][i + (hi ? 8 : 0)][n * 16 + lq] = (_Float16)h;
    }
  }

  const _Float16* hrow = &Hs[wave][lq][0];
#pragma unroll
  for (int n2 = 0; n2 < 4; ++n2) {
    v8f c = {};
#pragma unroll
    for (int k = 0; k < 4; ++k)
      c = wmma16(load_a_frag(hrow, k, hi),
                 load_b_frag(&W1T[n2 * 16 + lq][0], k, hi), c);
    const float bn = b1s[n2 * 16 + lq];
#pragma unroll
    for (int i = 0; i < 8; ++i) {
      const int row = wrow + i + (hi ? 8 : 0);
      ui_out[row * 64 + n2 * 16 + lq] = LREL(c[i] + bn);
    }
  }
}

// ------------------------------- AO kernel ---------------------------------
#define TOTROWS   (8192 * 32)
#define AO_BLOCKS 1024
#define AO_TILES  (TOTROWS / 64 / AO_BLOCKS)  // 4 row-tiles per block

__global__ __launch_bounds__(128) void ao_fused_kernel(
    const float* __restrict__ a_emb, const float* __restrict__ o_emb,
    const int* __restrict__ s,
    const float* __restrict__ W0, const float* __restrict__ b0,
    const float* __restrict__ W1, const float* __restrict__ b1,
    const float* __restrict__ ui, float* __restrict__ pred) {
  // All 3 relations resident for the whole block lifetime (~146 KB), plus
  // X tile + per-wave H buffers: ~178 KB of the 320 KB WGP pool.
  __shared__ __attribute__((aligned(32))) _Float16 W0T[3][128][128];  // 96 KB
  __shared__ __attribute__((aligned(32))) _Float16 W1T[3][64][128];   // 48 KB
  __shared__ __attribute__((aligned(32))) _Float16 Xs[64][128];       // 16 KB
  __shared__ __attribute__((aligned(32))) _Float16 Hs[4][16][128];    // 16 KB
  __shared__ float b0s[3 * 128];
  __shared__ float b1s[3 * 64];

  const int tid  = threadIdx.x;
  const int wave = tid >> 5;
  const int lane = tid & 31;
  const int lq   = lane & 15;
  const int hi   = lane >> 4;

  // ---- one-time weight stage: all 3 relations, transposed, f16 ----
  const float4* W0v = (const float4*)W0;
  for (int idx = tid; idx < (3 * 128 * 128) / 4; idx += 128) {
    const int e = idx * 4;
    const int r = e >> 14, rem = e & 16383;
    const int i = rem >> 7, o = rem & 127;
    const float4 w = W0v[idx];
    W0T[r][o + 0][i] = (_Float16)w.x;
    W0T[r][o + 1][i] = (_Float16)w.y;
    W0T[r][o + 2][i] = (_Float16)w.z;
    W0T[r][o + 3][i] = (_Float16)w.w;
  }
  const float4* W1v = (const float4*)W1;
  for (int idx = tid; idx < (3 * 128 * 64) / 4; idx += 128) {
    const int e = idx * 4;
    const int r = e >> 13, rem = e & 8191;
    const int i = rem >> 6, o = rem & 63;
    const float4 w = W1v[idx];
    W1T[r][o + 0][i] = (_Float16)w.x;
    W1T[r][o + 1][i] = (_Float16)w.y;
    W1T[r][o + 2][i] = (_Float16)w.z;
    W1T[r][o + 3][i] = (_Float16)w.w;
  }
  for (int idx = tid; idx < 3 * 128; idx += 128) b0s[idx] = b0[idx];
  if (tid < 3 * 64) b1s[tid] = b1[tid];

  const float4* a4 = (const float4*)a_emb;
  const float4* o4 = (const float4*)o_emb;

#pragma unroll 1
  for (int t = 0; t < AO_TILES; ++t) {
    const int rowBase = (blockIdx.x * AO_TILES + t) * 64;

    __syncthreads();  // Xs free to overwrite (also covers weight stage at t=0)
    for (int idx = tid; idx < 64 * 16; idx += 128) {
      const int row = idx >> 4, q = idx & 15;
      const int g = (rowBase + row) * 16 + q;
      *(v4h*)&Xs[row][q * 4]      = cvt4(a4[g]);
      *(v4h*)&Xs[row][64 + q * 4] = cvt4(o4[g]);
    }
    // prefetch next tile's rows (256 B each -> 2 lines) while we compute
    if (t + 1 < AO_TILES) {
      const int nrow = rowBase + 64 + (tid & 63);
      const float* p = (tid < 64 ? a_emb : o_emb) + nrow * 64;
      __builtin_prefetch(p, 0, 3);
      __builtin_prefetch(p + 32, 0, 3);
    }
    __syncthreads();

    const int wrow = rowBase + wave * 16;
    const int bIdx = wrow >> 5;  // one batch index per 16-row wave tile
    float uival[4];
#pragma unroll
    for (int n2 = 0; n2 < 4; ++n2) uival[n2] = ui[bIdx * 64 + n2 * 16 + lq];

    int srow = wrow + lq + (hi ? 8 : 0);
    if (srow >= TOTROWS) srow = TOTROWS - 1;  // clamp for never-storing lanes
    const int sval = s[srow];

    const _Float16* xrow = &Xs[wave * 16 + lq][0];
    v16h aX[4];
#pragma unroll
    for (int k = 0; k < 4; ++k) aX[k] = load_a_frag(xrow, k, hi);

#pragma unroll 1
    for (int r = 0; r < 3; ++r) {
      // layer 0: H_r = LReLU(X @ W0_r + b0_r) -> per-wave LDS (no barrier)
#pragma unroll
      for (int n = 0; n < 8; ++n) {
        v8f c = {};  // inline-0 SRC2 on first WMMA
#pragma unroll
        for (int k = 0; k < 4; ++k)
          c = wmma16(aX[k], load_b_frag(&W0T[r][n * 16 + lq][0], k, hi), c);
        const float bn = b0s[r * 128 + n * 16 + lq];
#pragma unroll
        for (int i = 0; i < 8; ++i) {
          const float h = LREL(c[i] + bn);
          Hs[wave][i + (hi ? 8 : 0)][n * 16 + lq] = (_Float16)h;
        }
      }

      // layer 1 fused with dot(ui, aos)
      const _Float16* hrow = &Hs[wave][lq][0];
      float partial[8] = {0.f, 0.f, 0.f, 0.f, 0.f, 0.f, 0.f, 0.f};
#pragma unroll
      for (int n2 = 0; n2 < 4; ++n2) {
        v8f c = {};
#pragma unroll
        for (int k = 0; k < 4; ++k)
          c = wmma16(load_a_frag(hrow, k, hi),
                     load_b_frag(&W1T[r][n2 * 16 + lq][0], k, hi), c);
        const float bn = b1s[r * 64 + n2 * 16 + lq];
#pragma unroll
        for (int i = 0; i < 8; ++i)
          partial[i] += LREL(c[i] + bn) * uival[n2];
      }

      // reduce across each 16-lane half (N dim); commit rows with s==r.
      // C layout: vgpr i -> M=i (lanes 0-15) / M=i+8 (lanes 16-31).
#pragma unroll
      for (int i = 0; i < 8; ++i) {
        float v = partial[i];
        v += __shfl_xor(v, 1, 32);
        v += __shfl_xor(v, 2, 32);
        v += __shfl_xor(v, 4, 32);
        v += __shfl_xor(v, 8, 32);
        if (lq == i && sval == r) pred[wrow + i + (hi ? 8 : 0)] = v;
      }
    }
  }
}

// ------------------------------ launcher -----------------------------------
extern "C" void kernel_launch(void* const* d_in, const int* in_sizes, int n_in,
                              void* d_out, int out_size, void* d_ws,
                              size_t ws_size, hipStream_t stream) {
  const float* u_emb = (const float*)d_in[0];
  const float* i_emb = (const float*)d_in[1];
  const float* a_emb = (const float*)d_in[2];
  const float* o_emb = (const float*)d_in[3];
  const int*   s     = (const int*)d_in[4];
  const float* ao_W0 = (const float*)d_in[5];
  const float* ao_b0 = (const float*)d_in[6];
  const float* ao_W1 = (const float*)d_in[7];
  const float* ao_b1 = (const float*)d_in[8];
  const float* ui_W0 = (const float*)d_in[9];
  const float* ui_b0 = (const float*)d_in[10];
  const float* ui_W1 = (const float*)d_in[11];
  const float* ui_b1 = (const float*)d_in[12];

  float* pred  = (float*)d_out;
  float* ui_ws = (float*)d_ws;  // 8192 * 64 floats = 2 MB

  ui_mlp_kernel<<<8192 / 64, 128, 0, stream>>>(u_emb, i_emb, ui_W0, ui_b0,
                                               ui_W1, ui_b1, ui_ws);
  ao_fused_kernel<<<AO_BLOCKS, 128, 0, stream>>>(
      a_emb, o_emb, s, ao_W0, ao_b0, ao_W1, ao_b1, ui_ws, pred);
}
